// BaseType_76768245449666
// MI455X (gfx1250) — matrix operand order repaired and checked by reading, not compiled, not run on hardware
//
#include <hip/hip_runtime.h>
#include <hip/hip_bf16.h>
#include <math.h>

typedef __attribute__((ext_vector_type(16))) _Float16 v16h;
typedef __attribute__((ext_vector_type(8)))  float    v8f;

__device__ __forceinline__ float sigf(float v) { return 1.0f / (1.0f + expf(-v)); }

// A fragment: reads p[0..7] and p[16..23] (16-bit A 16x32 layout, per-lane slice)
__device__ __forceinline__ v16h frag_a(const float* __restrict__ p) {
  const float4* q = (const float4*)p;
  float4 u0 = q[0], u1 = q[1], u2 = q[4], u3 = q[5];
  v16h a;
  a[0]  = (_Float16)u0.x; a[1]  = (_Float16)u0.y; a[2]  = (_Float16)u0.z; a[3]  = (_Float16)u0.w;
  a[4]  = (_Float16)u1.x; a[5]  = (_Float16)u1.y; a[6]  = (_Float16)u1.z; a[7]  = (_Float16)u1.w;
  a[8]  = (_Float16)u2.x; a[9]  = (_Float16)u2.y; a[10] = (_Float16)u2.z; a[11] = (_Float16)u2.w;
  a[12] = (_Float16)u3.x; a[13] = (_Float16)u3.y; a[14] = (_Float16)u3.z; a[15] = (_Float16)u3.w;
  return a;
}

// B fragment: reads p[0..7] and p[16..23] (dense 16-bit B 32x16 layout, vector broadcast to all cols)
__device__ __forceinline__ v16h frag_b(const float* __restrict__ p) {
  const float4* q = (const float4*)p;
  float4 u0 = q[0], u1 = q[1], u2 = q[4], u3 = q[5];
  v16h b;
  b[0]  = (_Float16)u0.x; b[1]  = (_Float16)u0.y; b[2]  = (_Float16)u0.z; b[3]  = (_Float16)u0.w;
  b[4]  = (_Float16)u1.x; b[5]  = (_Float16)u1.y; b[6]  = (_Float16)u1.z; b[7]  = (_Float16)u1.w;
  b[8]  = (_Float16)u2.x; b[9]  = (_Float16)u2.y; b[10] = (_Float16)u2.z; b[11] = (_Float16)u2.w;
  b[12] = (_Float16)u3.x; b[13] = (_Float16)u3.y; b[14] = (_Float16)u3.z; b[15] = (_Float16)u3.w;
  return b;
}

// ---------------------------------------------------------------------------
// WMMA GEMV: out[m] = sum_k W[m][k]*v[k] (+bias), 16 rows per wave32 block.
// Two independent accumulators, 64 K per iteration (2 WMMAs).
// ---------------------------------------------------------------------------
template <bool HAS_BIAS>
__global__ void gemv16_wmma(const float* __restrict__ W, const float* __restrict__ v,
                            const float* __restrict__ bias, float* __restrict__ out, int K) {
  const int lane = threadIdx.x;                 // 0..31
  const int m    = blockIdx.x * 16 + (lane & 15);
  const int kA   = (lane < 16) ? 0 : 8;         // A fragment K base (ISA layout)
  const int kB   = (lane < 16) ? 0 : 8;         // B fragment K base (ISA layout)
  const float* row = W + (size_t)m * K;
  v8f acc0 = {0.f,0.f,0.f,0.f,0.f,0.f,0.f,0.f};
  v8f acc1 = {0.f,0.f,0.f,0.f,0.f,0.f,0.f,0.f};
  for (int k0 = 0; k0 < K; k0 += 64) {
    v16h a0 = frag_a(row + k0 + kA);
    v16h b0 = frag_b(v   + k0 + kB);
    v16h a1 = frag_a(row + k0 + 32 + kA);
    v16h b1 = frag_b(v   + k0 + 32 + kB);
    acc0 = __builtin_amdgcn_wmma_f32_16x16x32_f16(false, a0, false, b0, (short)0, acc0, false, false);
    acc1 = __builtin_amdgcn_wmma_f32_16x16x32_f16(false, a1, false, b1, (short)0, acc1, false, false);
  }
  v8f acc = acc0 + acc1;
  // D: lane 0 -> rows 0..7 (N=0), lane 16 -> rows 8..15 (N=0)
  if ((lane & 15) == 0) {
    const int base = blockIdx.x * 16 + ((lane < 16) ? 0 : 8);
    float4 r0, r1;
    if (HAS_BIAS) {
      r0 = make_float4(acc[0] + bias[base + 0], acc[1] + bias[base + 1],
                       acc[2] + bias[base + 2], acc[3] + bias[base + 3]);
      r1 = make_float4(acc[4] + bias[base + 4], acc[5] + bias[base + 5],
                       acc[6] + bias[base + 6], acc[7] + bias[base + 7]);
    } else {
      r0 = make_float4(acc[0], acc[1], acc[2], acc[3]);
      r1 = make_float4(acc[4], acc[5], acc[6], acc[7]);
    }
    float4* o = (float4*)(out + base);
    o[0] = r0; o[1] = r1;
  }
}

// gates = W_ih·x + W_hh·h0 + b_ih + b_hh  (2048 rows, two K=512 passes)
__global__ void gates_wmma(const float* __restrict__ Wih, const float* __restrict__ Whh,
                           const float* __restrict__ x, const float* __restrict__ h0,
                           const float* __restrict__ bih, const float* __restrict__ bhh,
                           float* __restrict__ gates) {
  const int lane = threadIdx.x;
  const int m    = blockIdx.x * 16 + (lane & 15);
  const int kA   = (lane < 16) ? 0 : 8;
  const int kB   = (lane < 16) ? 0 : 8;
  const float* rowi = Wih + (size_t)m * 512;
  const float* rowh = Whh + (size_t)m * 512;
  v8f acc0 = {0.f,0.f,0.f,0.f,0.f,0.f,0.f,0.f};
  v8f acc1 = {0.f,0.f,0.f,0.f,0.f,0.f,0.f,0.f};
  for (int k0 = 0; k0 < 512; k0 += 64) {
    v16h a0 = frag_a(rowi + k0 + kA);
    v16h b0 = frag_b(x    + k0 + kB);
    v16h a1 = frag_a(rowi + k0 + 32 + kA);
    v16h b1 = frag_b(x    + k0 + 32 + kB);
    acc0 = __builtin_amdgcn_wmma_f32_16x16x32_f16(false, a0, false, b0, (short)0, acc0, false, false);
    acc1 = __builtin_amdgcn_wmma_f32_16x16x32_f16(false, a1, false, b1, (short)0, acc1, false, false);
  }
  for (int k0 = 0; k0 < 512; k0 += 64) {
    v16h a0 = frag_a(rowh + k0 + kA);
    v16h b0 = frag_b(h0   + k0 + kB);
    v16h a1 = frag_a(rowh + k0 + 32 + kA);
    v16h b1 = frag_b(h0   + k0 + 32 + kB);
    acc0 = __builtin_amdgcn_wmma_f32_16x16x32_f16(false, a0, false, b0, (short)0, acc0, false, false);
    acc1 = __builtin_amdgcn_wmma_f32_16x16x32_f16(false, a1, false, b1, (short)0, acc1, false, false);
  }
  v8f acc = acc0 + acc1;
  if ((lane & 15) == 0) {
    const int base = blockIdx.x * 16 + ((lane < 16) ? 0 : 8);
    float4 r0 = make_float4(acc[0] + bih[base + 0] + bhh[base + 0],
                            acc[1] + bih[base + 1] + bhh[base + 1],
                            acc[2] + bih[base + 2] + bhh[base + 2],
                            acc[3] + bih[base + 3] + bhh[base + 3]);
    float4 r1 = make_float4(acc[4] + bih[base + 4] + bhh[base + 4],
                            acc[5] + bih[base + 5] + bhh[base + 5],
                            acc[6] + bih[base + 6] + bhh[base + 6],
                            acc[7] + bih[base + 7] + bhh[base + 7]);
    float4* o = (float4*)(gates + base);
    o[0] = r0; o[1] = r1;
  }
}

// ---------------------------------------------------------------------------
// Big streaming GEMV (W_out, 102 MB): one wave32 per row, float4 loads.
// ---------------------------------------------------------------------------
__global__ void gemv_rows_f32(const float* __restrict__ W, const float* __restrict__ v,
                              const float* __restrict__ bias, float* __restrict__ out,
                              int rows, int K) {
  const int gtid = blockIdx.x * blockDim.x + threadIdx.x;
  const int row  = gtid >> 5;
  const int lane = gtid & 31;
  if (row >= rows) return;
  const float4* Wr = (const float4*)(W + (size_t)row * K);
  const float4* V4 = (const float4*)v;
  float s = 0.0f;
  const int n4 = K >> 2;
#pragma unroll 4
  for (int i = lane; i < n4; i += 32) {
    float4 a = Wr[i], b = V4[i];
    s += a.x * b.x + a.y * b.y + a.z * b.z + a.w * b.w;
  }
#pragma unroll
  for (int off = 16; off > 0; off >>= 1) s += __shfl_down(s, off, 32);
  if (lane == 0) out[row] = s + bias[row];
}

// ---------------------------------------------------------------------------
// LSTM cell elementwise: c = sig(f)*c0 + sig(i)*tanh(g); h = sig(o)*tanh(c)
// ---------------------------------------------------------------------------
__global__ void lstm_cell(const float* __restrict__ gates, const float* __restrict__ c0,
                          float* __restrict__ h_ws, float* __restrict__ out_h,
                          float* __restrict__ out_c) {
  const int j = blockIdx.x * blockDim.x + threadIdx.x;
  if (j >= 512) return;
  float ig = sigf(gates[j]);
  float fg = sigf(gates[512 + j]);
  float gg = tanhf(gates[1024 + j]);
  float og = sigf(gates[1536 + j]);
  float c  = fg * c0[j] + ig * gg;
  float h  = og * tanhf(c);
  h_ws[j] = h; out_h[j] = h; out_c[j] = c;
}

// gen_p = sigmoid(W_gen · [h, name_hidden, type_hidden, x] + b_gen)
__global__ void genp_kernel(const float* __restrict__ h, const float* __restrict__ nameh,
                            const float* __restrict__ typeh, const float* __restrict__ x,
                            const float* __restrict__ Wgen, const float* __restrict__ bgen,
                            float* __restrict__ genp) {
  __shared__ float red[256];
  const int tid = threadIdx.x;
  float s = 0.0f;
  for (int idx = tid; idx < 2048; idx += 256) {
    float val = (idx < 512)  ? h[idx]
              : (idx < 1024) ? nameh[idx - 512]
              : (idx < 1536) ? typeh[idx - 1024]
                             : x[idx - 1536];
    s += val * Wgen[idx];
  }
  red[tid] = s; __syncthreads();
  for (int st = 128; st > 0; st >>= 1) {
    if (tid < st) red[tid] += red[tid + st];
    __syncthreads();
  }
  if (tid == 0) genp[0] = sigf(red[0] + bgen[0]);
}

__global__ void zero_kernel(float* __restrict__ p, int n) {
  const int i = blockIdx.x * blockDim.x + threadIdx.x;
  if (i < n) p[i] = 0.0f;
}

// softmax(scores) then scatter-add into p_copy[type_indices[t]]
__global__ void attn_scatter(const float* __restrict__ scores, const int* __restrict__ idx,
                             float* __restrict__ p_copy, int T) {
  __shared__ float red[1024];
  const int tid = threadIdx.x;
  float m = -3.4e38f;
  for (int t = tid; t < T; t += 1024) m = fmaxf(m, scores[t]);
  red[tid] = m; __syncthreads();
  for (int st = 512; st > 0; st >>= 1) {
    if (tid < st) red[tid] = fmaxf(red[tid], red[tid + st]);
    __syncthreads();
  }
  const float smax = red[0]; __syncthreads();
  float s = 0.0f;
  for (int t = tid; t < T; t += 1024) s += expf(scores[t] - smax);
  red[tid] = s; __syncthreads();
  for (int st = 512; st > 0; st >>= 1) {
    if (tid < st) red[tid] += red[tid + st];
    __syncthreads();
  }
  const float inv = 1.0f / red[0];
  for (int t = tid; t < T; t += 1024)
    atomicAdd(&p_copy[idx[t]], expf(scores[t] - smax) * inv);
}

// global max + sum(exp) over logits (single block, grid-stride)
__global__ void softmax_stats(const float* __restrict__ logits, float* __restrict__ stats, int n) {
  __shared__ float red[1024];
  const int tid = threadIdx.x;
  float m = -3.4e38f;
  for (int i = tid; i < n; i += 1024) m = fmaxf(m, logits[i]);
  red[tid] = m; __syncthreads();
  for (int st = 512; st > 0; st >>= 1) {
    if (tid < st) red[tid] = fmaxf(red[tid], red[tid + st]);
    __syncthreads();
  }
  const float smax = red[0]; __syncthreads();
  float s = 0.0f;
  for (int i = tid; i < n; i += 1024) s += expf(logits[i] - smax);
  red[tid] = s; __syncthreads();
  for (int st = 512; st > 0; st >>= 1) {
    if (tid < st) red[tid] += red[tid + st];
    __syncthreads();
  }
  if (tid == 0) { stats[0] = smax; stats[1] = red[0]; }
}

// log_prob[v] = log(max(gen_p*p_vocab[v] + (1-gen_p)*p_copy[v], 1e-10))
__global__ void final_prob(const float* __restrict__ logits, const float* __restrict__ stats,
                           const float* __restrict__ p_copy, const float* __restrict__ genp,
                           float* __restrict__ out, int V, int total) {
  const int v = blockIdx.x * blockDim.x + threadIdx.x;
  if (v >= total) return;
  const float gp = genp[0];
  const float pv = (v < V) ? expf(logits[v] - stats[0]) / stats[1] : 0.0f;
  const float prob = gp * pv + (1.0f - gp) * p_copy[v];
  out[v] = logf(fmaxf(prob, 1e-10f));
}

extern "C" void kernel_launch(void* const* d_in, const int* in_sizes, int n_in,
                              void* d_out, int out_size, void* d_ws, size_t ws_size,
                              hipStream_t stream) {
  const float* x       = (const float*)d_in[0];
  const float* h0      = (const float*)d_in[1];
  const float* c0      = (const float*)d_in[2];
  const float* nameh   = (const float*)d_in[3];
  const float* typeh   = (const float*)d_in[4];
  const float* typehs  = (const float*)d_in[5];   // 4096 x 512
  const int*   typeidx = (const int*)d_in[6];     // 4096 int32
  // d_in[7] = max_oov (constant 20)
  const float* Wih   = (const float*)d_in[8];     // 2048 x 512
  const float* Whh   = (const float*)d_in[9];     // 2048 x 512
  const float* bih   = (const float*)d_in[10];
  const float* bhh   = (const float*)d_in[11];
  const float* Wattn = (const float*)d_in[12];    // 512 x 512
  const float* battn = (const float*)d_in[13];
  const float* Wgen  = (const float*)d_in[14];    // 1 x 2048
  const float* bgen  = (const float*)d_in[15];
  const float* Wout  = (const float*)d_in[16];    // 50000 x 512
  const float* bout  = (const float*)d_in[17];

  float* ws     = (float*)d_ws;
  float* GATES  = ws;              // 2048
  float* HVEC   = ws + 2048;       // 512
  float* PROJ   = ws + 2560;       // 512
  float* SCORES = ws + 3072;       // 4096
  float* LOGITS = ws + 7168;       // 50000
  float* PCOPY  = ws + 57168;      // 50020
  float* STATS  = ws + 107188;     // 2 (max, sum)
  float* GENP   = ws + 107190;     // 1

  float* out      = (float*)d_out;
  float* out_logp = out;           // 50020
  float* out_h    = out + 50020;   // 512
  float* out_c    = out + 50532;   // 512

  zero_kernel<<<dim3(196), dim3(256), 0, stream>>>(PCOPY, 50020);
  gates_wmma<<<dim3(128), dim3(32), 0, stream>>>(Wih, Whh, x, h0, bih, bhh, GATES);
  lstm_cell<<<dim3(2), dim3(256), 0, stream>>>(GATES, c0, HVEC, out_h, out_c);
  genp_kernel<<<dim3(1), dim3(256), 0, stream>>>(HVEC, nameh, typeh, x, Wgen, bgen, GENP);
  gemv16_wmma<true><<<dim3(32), dim3(32), 0, stream>>>(Wattn, HVEC, battn, PROJ, 512);
  gemv16_wmma<false><<<dim3(256), dim3(32), 0, stream>>>(typehs, PROJ, nullptr, SCORES, 512);
  gemv_rows_f32<<<dim3(6250), dim3(256), 0, stream>>>(Wout, HVEC, bout, LOGITS, 50000, 512);
  attn_scatter<<<dim3(1), dim3(1024), 0, stream>>>(SCORES, typeidx, PCOPY, 4096);
  softmax_stats<<<dim3(1), dim3(1024), 0, stream>>>(LOGITS, STATS, 50000);
  final_prob<<<dim3(196), dim3(256), 0, stream>>>(LOGITS, STATS, PCOPY, GENP, out_logp, 50000, 50020);
}
